// TNetwork_17454747091450
// MI455X (gfx1250) — matrix operand order repaired and checked by reading, not compiled, not run on hardware
//
#include <hip/hip_runtime.h>

// ---------------------------------------------------------------------------
// GCN forward for MI455X (gfx1250, wave32).
//  - Dense GEMMs via v_wmma_f32_16x16x32_bf16 (fp32 accumulate).
//  - Weights pre-transposed+converted to bf16 once ([Nc][K] layout); the
//    128xK bf16 weight panel is staged into LDS per block by the Tensor Data
//    Mover (tensor_load_to_lds + s_wait_tensorcnt), falling back to a
//    cooperative VGPR copy if the TDM builtin is unavailable.
//  - B-fragments are two 16B runs read back with ds_load_b128.
//  - A stream stays fp32 (float4 loads + v_cvt_pk_bf16_f32), prefetched with
//    global_prefetch_b8.
//  - Edge message passing via float4 gathers + global_atomic_add_f32 scatter.
// ---------------------------------------------------------------------------

typedef __attribute__((ext_vector_type(16))) __bf16 v16bf;
typedef __attribute__((ext_vector_type(8)))  __bf16 v8bf;
typedef __attribute__((ext_vector_type(8)))  float  v8f;
typedef unsigned int u32x4 __attribute__((ext_vector_type(4)));
typedef int          i32x8 __attribute__((ext_vector_type(8)));
typedef int          i32x4 __attribute__((ext_vector_type(4)));

#define HCH    128
#define NGRAPH 64
#define OUTC   10

#if defined(__HIP_DEVICE_COMPILE__) && \
    __has_builtin(__builtin_amdgcn_tensor_load_to_lds) && \
    __has_builtin(__builtin_amdgcn_s_wait_tensorcnt)
#define GCN_USE_TDM 1
#else
#define GCN_USE_TDM 0
#endif

// ---------------------------------------------------------------------------
// Weight prep: W[K x Nc] fp32 row-major  ->  Wt[Nc x K] bf16 row-major
// ---------------------------------------------------------------------------
__global__ __launch_bounds__(256)
void k_wt(const float* __restrict__ W, __bf16* __restrict__ Wt, int K, int Nc)
{
    int t = blockIdx.x * blockDim.x + threadIdx.x;
    if (t >= K * Nc) return;
    int k = t / Nc;
    int n = t - k * Nc;
    Wt[(size_t)n * K + k] = (__bf16)W[t];
}

// ---------------------------------------------------------------------------
// C[MxNc] = op(A[MxK]) @ W[KxNc] (+bias) (opt. ReLU out), bf16 WMMA.
// Wt is the bf16 [Nc x K] transposed weight. Requires M%16==0, K%32==0,
// Nc%128==0. Block = 128 threads = 4 waves; each wave -> 16 rows x 128 cols.
// grid.x = ceil(M/64), grid.y = Nc/128. Dynamic LDS = 256*K bytes.
// ---------------------------------------------------------------------------
__global__ __launch_bounds__(128)
void k_gemm_bf16(const float* __restrict__ A, const __bf16* __restrict__ Wt,
                 const float* __restrict__ bias, float* __restrict__ C,
                 int M, int K, int Nc, int reluIn, int reluOut)
{
    extern __shared__ __bf16 sW[];            // [128 cols][K] panel, col-contig
    const int lane   = threadIdx.x & 31;
    const int wave   = threadIdx.x >> 5;
    const int nChunk = blockIdx.y * 128;

#if GCN_USE_TDM
    // ---- panel load via Tensor Data Mover: one 2D-tile DMA per block ----
    // D# per CDNA5 ISA 8.3/8.4: 2D tile, dim0 = K halves (contiguous),
    // dim1 = 128 rows (cols of W), packed contiguously into LDS.
    if (wave == 0) {
        unsigned long long ga =
            (unsigned long long)(size_t)(const void*)(Wt + (size_t)nChunk * (size_t)K);
        unsigned lds = (unsigned)(size_t)(void*)sW;   // flat low 32 bits = LDS offset

        u32x4 g0;
        g0[0] = 1u;                                   // count=1, no gather
        g0[1] = lds;                                  // lds_addr (bytes)
        g0[2] = (unsigned)(ga & 0xFFFFFFFFu);         // global_addr[31:0]
        g0[3] = (unsigned)((ga >> 32) & 0x01FFFFFFu)  // global_addr[56:32]
              | (2u << 30);                           // type = 2 ("image")

        i32x8 g1;
        g1[0] = (int)(1u << 16);                      // data_size=1 (2 bytes)
        g1[1] = (int)((unsigned)(K & 0xFFFF) << 16);  // tensor_dim0[15:0]
        g1[2] = (int)(((unsigned)K >> 16) |           // tensor_dim0[31:16]
                      (128u << 16));                  // tensor_dim1[15:0]=128
        g1[3] = (int)((unsigned)(K & 0xFFFF) << 16);  // tile_dim0 = K
        g1[4] = (int)128u;                            // tile_dim1=128, dim2=0
        g1[5] = (int)(unsigned)K;                     // dim0_stride[31:0] = K
        g1[6] = 0;                                    // dim0_stride hi, dim1_stride lo
        g1[7] = 0;                                    // dim1_stride hi (2D: unused)

        i32x4 gz4 = {0, 0, 0, 0};                     // groups 2/3: 2D tensor
        i32x8 gz8 = {0, 0, 0, 0, 0, 0, 0, 0};         // trailing group (unused)
        __builtin_amdgcn_tensor_load_to_lds(g0, g1, gz4, gz4, gz8, 0);
    }
    __builtin_amdgcn_s_wait_tensorcnt(0);
#else
    // ---- fallback: cooperative panel load through VGPRs ----
    {
        const __bf16* gW = Wt + (size_t)nChunk * (size_t)K;
        const int totalChunks = (128 * K) >> 3;        // v8bf chunks
        for (int ci = threadIdx.x; ci < totalChunks; ci += 128) {
            *reinterpret_cast<v8bf*>(sW + ((size_t)ci << 3)) =
                *reinterpret_cast<const v8bf*>(gW + ((size_t)ci << 3));
        }
    }
#endif
    __syncthreads();

    const int mBase = (blockIdx.x * 4 + wave) * 16;
    if (mBase >= M) return;                   // whole-wave guard, after barrier

    const int mRow  = mBase + (lane & 15);    // A: row per lane
    const int aKoff = (lane & 16) ? 8 : 0;    // A: lanes16-31 hold K+8 / K+24
    const int nFrag = lane & 15;              // B: column-in-tile per lane
    const int bKoff = (lane & 16) ? 8 : 0;    // B: lanes16-31 hold K+8 / K+24

    v8f acc[8];
#pragma unroll
    for (int t = 0; t < 8; ++t)
#pragma unroll
        for (int r = 0; r < 8; ++r) acc[t][r] = 0.0f;

    const float* Arow = A + (size_t)mRow * (size_t)K;

    for (int k0 = 0; k0 < K; k0 += 32) {
        // prefetch next A chunk into cache while this step computes
        __builtin_prefetch(Arow + k0 + 32, 0, 1);

        // ---- A fragment: two contiguous 8-float runs -> 4x float4 loads ----
        const float* pa = Arow + k0 + aKoff;
        float4 x0 = *reinterpret_cast<const float4*>(pa + 0);
        float4 x1 = *reinterpret_cast<const float4*>(pa + 4);
        float4 x2 = *reinterpret_cast<const float4*>(pa + 16);
        float4 x3 = *reinterpret_cast<const float4*>(pa + 20);
        float av[16] = { x0.x, x0.y, x0.z, x0.w,  x1.x, x1.y, x1.z, x1.w,
                         x2.x, x2.y, x2.z, x2.w,  x3.x, x3.y, x3.z, x3.w };
        if (reluIn) {
#pragma unroll
            for (int i = 0; i < 16; ++i) av[i] = av[i] > 0.f ? av[i] : 0.f;
        }
        v16bf a;
#pragma unroll
        for (int i = 0; i < 16; ++i) a[i] = (__bf16)av[i];

        // ---- 8 column tiles; B fragment = two 16-byte runs from LDS ----
#pragma unroll
        for (int t = 0; t < 8; ++t) {
            const __bf16* bp = sW + (size_t)(t * 16 + nFrag) * (size_t)K
                                  + k0 + bKoff;
            v8bf blo = *reinterpret_cast<const v8bf*>(bp);
            v8bf bhi = *reinterpret_cast<const v8bf*>(bp + 16);
            v16bf b;
#pragma unroll
            for (int i = 0; i < 8; ++i) { b[i] = blo[i]; b[i + 8] = bhi[i]; }
            acc[t] = __builtin_amdgcn_wmma_f32_16x16x32_bf16(
                false, a, false, b, (short)0, acc[t], false, false);
        }
    }

    // ---- epilogue: v8f elem r -> row r + 8*(lane>=16), col = lane&15 ----
    const int rowOff = (lane >> 4) * 8;
#pragma unroll
    for (int t = 0; t < 8; ++t) {
        const int col = nChunk + t * 16 + (lane & 15);
        const float bv = bias ? bias[col] : 0.0f;
#pragma unroll
        for (int r = 0; r < 8; ++r) {
            float v = acc[t][r] + bv;
            if (reluOut) v = v > 0.f ? v : 0.f;
            C[(size_t)(mBase + rowOff + r) * (size_t)Nc + col] = v;
        }
    }
}

// ---------------------------------------------------------------------------
// Small helper kernels
// ---------------------------------------------------------------------------
__global__ __launch_bounds__(256) void k_zero(float* p, int n)
{
    int t = blockIdx.x * blockDim.x + threadIdx.x;
    if (t < n) p[t] = 0.0f;
}

__global__ __launch_bounds__(256) void k_deg(const int* __restrict__ ei,
                                             float* __restrict__ deg, int E)
{
    int e = blockIdx.x * blockDim.x + threadIdx.x;
    if (e < E) atomicAdd(&deg[ei[E + e]], 1.0f);
}

__global__ __launch_bounds__(256) void k_dinv(const float* __restrict__ deg,
                                              float* __restrict__ dinv, int N)
{
    int i = blockIdx.x * blockDim.x + threadIdx.x;
    if (i < N) dinv[i] = rsqrtf(deg[i] + 1.0f);   // deg+1 >= 1 always
}

// out[i,:] = xw[i,:] * dinv[i]^2 + bias  (self-loop term initializes the sum)
__global__ __launch_bounds__(256) void k_self_init(const float* __restrict__ xw,
                                                   const float* __restrict__ dinv,
                                                   const float* __restrict__ bias,
                                                   float* __restrict__ out, int N)
{
    int t = blockIdx.x * blockDim.x + threadIdx.x;
    if (t >= N * 32) return;
    int i = t >> 5;
    int c = (t & 31) << 2;
    float s = dinv[i]; s = s * s;
    float4 v = *reinterpret_cast<const float4*>(xw + (size_t)i * HCH + c);
    float4 b = *reinterpret_cast<const float4*>(bias + c);
    float4 r; r.x = v.x * s + b.x; r.y = v.y * s + b.y;
              r.z = v.z * s + b.z; r.w = v.w * s + b.w;
    *reinterpret_cast<float4*>(out + (size_t)i * HCH + c) = r;
}

// out[dst,:] += xw[src,:] * (dinv[src]*dinv[dst])  for every edge
__global__ __launch_bounds__(256) void k_edge_scatter(const int* __restrict__ ei,
                                                      const float* __restrict__ xw,
                                                      const float* __restrict__ dinv,
                                                      float* __restrict__ out, int E)
{
    int t = blockIdx.x * blockDim.x + threadIdx.x;
    if (t >= E * 32) return;
    int e = t >> 5;
    int c = (t & 31) << 2;
    int s = ei[e];
    int d = ei[E + e];
    float w = dinv[s] * dinv[d];
    float4 v = *reinterpret_cast<const float4*>(xw + (size_t)s * HCH + c);
    float* o = out + (size_t)d * HCH + c;
    atomicAdd(o + 0, v.x * w);
    atomicAdd(o + 1, v.y * w);
    atomicAdd(o + 2, v.z * w);
    atomicAdd(o + 3, v.w * w);
}

__global__ __launch_bounds__(256) void k_pool_acc(const float* __restrict__ h,
                                                  const int* __restrict__ batch,
                                                  float* __restrict__ sums,
                                                  float* __restrict__ cnt, int N)
{
    int t = blockIdx.x * blockDim.x + threadIdx.x;
    if (t >= N * 32) return;
    int i = t >> 5;
    int c = (t & 31) << 2;
    int g = batch[i];
    float4 v = *reinterpret_cast<const float4*>(h + (size_t)i * HCH + c);
    float* o = sums + (size_t)g * HCH + c;
    atomicAdd(o + 0, v.x);
    atomicAdd(o + 1, v.y);
    atomicAdd(o + 2, v.z);
    atomicAdd(o + 3, v.w);
    if ((t & 31) == 0) atomicAdd(&cnt[g], 1.0f);
}

__global__ __launch_bounds__(256) void k_pool_div(const float* __restrict__ sums,
                                                  const float* __restrict__ cnt,
                                                  float* __restrict__ f,
                                                  float* __restrict__ outF)
{
    int t = blockIdx.x * blockDim.x + threadIdx.x;
    if (t >= NGRAPH * HCH) return;
    int g = t >> 7;
    float c = cnt[g];
    float v = sums[t] / (c > 1.0f ? c : 1.0f);
    f[t] = v;
    outF[t] = v;
}

// tiny 128 -> 10 head (not WMMA-shaped; scalar dot products)
__global__ __launch_bounds__(256) void k_fc5(const float* __restrict__ y4,
                                             const float* __restrict__ w,
                                             const float* __restrict__ b,
                                             float* __restrict__ out)
{
    int t = blockIdx.x * blockDim.x + threadIdx.x;
    if (t >= NGRAPH * OUTC) return;
    int g = t / OUTC;
    int o = t % OUTC;
    float acc = b[o];
    const float* yg = y4 + (size_t)g * HCH;
#pragma unroll 4
    for (int k = 0; k < HCH; ++k) acc += yg[k] * w[(size_t)k * OUTC + o];
    out[t] = acc;
}

// ---------------------------------------------------------------------------
// Host orchestration
// ---------------------------------------------------------------------------
extern "C" void kernel_launch(void* const* d_in, const int* in_sizes, int n_in,
                              void* d_out, int out_size, void* d_ws, size_t ws_size,
                              hipStream_t stream)
{
    (void)n_in; (void)out_size; (void)ws_size;

    const float* x     = (const float*)d_in[0];
    const int*   ei    = (const int*)  d_in[1];
    const int*   batch = (const int*)  d_in[2];
    const float* w1 = (const float*)d_in[3],  *b1 = (const float*)d_in[4];
    const float* w2 = (const float*)d_in[5],  *b2 = (const float*)d_in[6];
    const float* w3 = (const float*)d_in[7],  *b3 = (const float*)d_in[8];
    const float* f1w = (const float*)d_in[9],  *f1b = (const float*)d_in[10];
    const float* f2w = (const float*)d_in[11], *f2b = (const float*)d_in[12];
    const float* f3w = (const float*)d_in[13], *f3b = (const float*)d_in[14];
    const float* f4w = (const float*)d_in[15], *f4b = (const float*)d_in[16];
    const float* f5w = (const float*)d_in[17], *f5b = (const float*)d_in[18];

    const int N = in_sizes[0] / HCH;     // 50000
    const int E = in_sizes[1] / 2;       // 800000

    // byte-based workspace carve with 64B alignment
    char* wp = (char*)d_ws;
    auto alloc = [&](size_t bytes) -> void* {
        void* r = (void*)wp;
        wp += (bytes + 63) & ~(size_t)63;
        return r;
    };
    float*  deg  = (float*) alloc((size_t)N * 4);
    float*  dinv = (float*) alloc((size_t)N * 4);
    float*  xw   = (float*) alloc((size_t)N * HCH * 4);
    float*  h0   = (float*) alloc((size_t)N * HCH * 4);
    float*  h1   = (float*) alloc((size_t)N * HCH * 4);
    float*  sums = (float*) alloc(NGRAPH * HCH * 4);
    float*  cnt  = (float*) alloc(NGRAPH * 4);
    float*  f    = (float*) alloc(NGRAPH * HCH * 4);
    float*  y1   = (float*) alloc(NGRAPH * 8 * HCH * 4);
    float*  y2   = (float*) alloc(NGRAPH * 4 * HCH * 4);
    float*  y3   = (float*) alloc(NGRAPH * 2 * HCH * 4);
    float*  y4   = (float*) alloc(NGRAPH * HCH * 4);
    __bf16* wt1  = (__bf16*)alloc(128  * 128  * 2);
    __bf16* wt2  = (__bf16*)alloc(128  * 128  * 2);
    __bf16* wt3  = (__bf16*)alloc(128  * 128  * 2);
    __bf16* wtf1 = (__bf16*)alloc(128  * 1024 * 2);
    __bf16* wtf2 = (__bf16*)alloc(1024 * 512  * 2);
    __bf16* wtf3 = (__bf16*)alloc(512  * 256  * 2);
    __bf16* wtf4 = (__bf16*)alloc(256  * 128  * 2);

    float* outF = (float*)d_out;               // [64,128]
    float* outY = (float*)d_out + NGRAPH*HCH;  // [64,10]

    const int T = 256;
    auto cdiv = [](long long a, long long b) { return (unsigned)((a + b - 1) / b); };

    // ---- weight prep: fp32 [KxNc] -> bf16 [NcxK] ----
    k_wt<<<cdiv(128  * 128,  T), T, 0, stream>>>(w1,  wt1,  128,  128);
    k_wt<<<cdiv(128  * 128,  T), T, 0, stream>>>(w2,  wt2,  128,  128);
    k_wt<<<cdiv(128  * 128,  T), T, 0, stream>>>(w3,  wt3,  128,  128);
    k_wt<<<cdiv(128  * 1024, T), T, 0, stream>>>(f1w, wtf1, 128,  1024);
    k_wt<<<cdiv(1024 * 512,  T), T, 0, stream>>>(f2w, wtf2, 1024, 512);
    k_wt<<<cdiv(512  * 256,  T), T, 0, stream>>>(f3w, wtf3, 512,  256);
    k_wt<<<cdiv(256  * 128,  T), T, 0, stream>>>(f4w, wtf4, 256,  128);

    // ---- degrees + symmetric norm factor ----
    k_zero<<<cdiv(N, T), T, 0, stream>>>(deg, N);
    k_deg <<<cdiv(E, T), T, 0, stream>>>(ei, deg, E);
    k_dinv<<<cdiv(N, T), T, 0, stream>>>(deg, dinv, N);

    const dim3 gconv(cdiv(N, 64), HCH / 128);
    const unsigned nodeT = cdiv((long long)N * 32, T);
    const unsigned edgeT = cdiv((long long)E * 32, T);
    const size_t lds128 = 256 * 128;           // 256*K bytes of panel LDS

    // conv1
    k_gemm_bf16<<<gconv, 128, lds128, stream>>>(x,  wt1, nullptr, xw, N, HCH, HCH, 0, 0);
    k_self_init<<<nodeT, T, 0, stream>>>(xw, dinv, b1, h0, N);
    k_edge_scatter<<<edgeT, T, 0, stream>>>(ei, xw, dinv, h0, E);
    // conv2 (ReLU of h0 fused into A-load)
    k_gemm_bf16<<<gconv, 128, lds128, stream>>>(h0, wt2, nullptr, xw, N, HCH, HCH, 1, 0);
    k_self_init<<<nodeT, T, 0, stream>>>(xw, dinv, b2, h1, N);
    k_edge_scatter<<<edgeT, T, 0, stream>>>(ei, xw, dinv, h1, E);
    // conv3 (no ReLU on output)
    k_gemm_bf16<<<gconv, 128, lds128, stream>>>(h1, wt3, nullptr, xw, N, HCH, HCH, 1, 0);
    k_self_init<<<nodeT, T, 0, stream>>>(xw, dinv, b3, h0, N);
    k_edge_scatter<<<edgeT, T, 0, stream>>>(ei, xw, dinv, h0, E);

    // global mean pool
    k_zero<<<cdiv(NGRAPH * HCH, T), T, 0, stream>>>(sums, NGRAPH * HCH);
    k_zero<<<1, 64, 0, stream>>>(cnt, NGRAPH);
    k_pool_acc<<<nodeT, T, 0, stream>>>(h0, batch, sums, cnt, N);
    k_pool_div<<<cdiv(NGRAPH * HCH, T), T, 0, stream>>>(sums, cnt, f, outF);

    // fc stack (M=64, WMMA, bias+ReLU fused in epilogue); LDS = 256*K bytes
    k_gemm_bf16<<<dim3(1, 8), 128, 256 * 128,  stream>>>(f,  wtf1, f1b, y1, NGRAPH, 128,  1024, 0, 1);
    k_gemm_bf16<<<dim3(1, 4), 128, 256 * 1024, stream>>>(y1, wtf2, f2b, y2, NGRAPH, 1024, 512,  0, 1);
    k_gemm_bf16<<<dim3(1, 2), 128, 256 * 512,  stream>>>(y2, wtf3, f3b, y3, NGRAPH, 512,  256,  0, 1);
    k_gemm_bf16<<<dim3(1, 1), 128, 256 * 256,  stream>>>(y3, wtf4, f4b, y4, NGRAPH, 256,  128,  0, 1);
    k_fc5<<<cdiv(NGRAPH * OUTC, T), T, 0, stream>>>(y4, f5w, f5b, outY);
}
